// MoE_60498909331510
// MI455X (gfx1250) — matrix operand order repaired and checked by reading, not compiled, not run on hardware
//
#include <hip/hip_runtime.h>
#include <hip/hip_bf16.h>
#include <math.h>

// ---------------- problem constants ----------------
#define D_MODEL 1024
#define FFN     4096
#define N_EXP   8
#define BS      4
#define SEQ     2048
#define NTOK    (BS * SEQ)           // 8192

// ---------------- WMMA types (gfx1250, wave32) ----------------
typedef __bf16 v16bf __attribute__((ext_vector_type(16)));
typedef float  v8f   __attribute__((ext_vector_type(8)));

union FragU { v16bf v; uint4 u[2]; };

__device__ __forceinline__ v8f wmma_bf16(v16bf a, v16bf b, v8f c) {
    // D = A(16x32 bf16) * B(32x16 bf16) + C(16x16 f32)
    return __builtin_amdgcn_wmma_f32_16x16x32_bf16(
        /*neg_a=*/false, a, /*neg_b=*/false, b,
        /*c_mod=*/(short)0, c, /*reuse_a=*/false, /*reuse_b=*/false);
}

__device__ __forceinline__ unsigned short f2bf(float f) {
    unsigned int u = __float_as_uint(f);
    u += 0x7FFFu + ((u >> 16) & 1u);      // round-to-nearest-even
    return (unsigned short)(u >> 16);
}

__device__ __forceinline__ float gelu_exact(float v) {
    return 0.5f * v * (1.0f + erff(v * 0.70710678118f));
}

// ---------------- 1) elementwise f32 -> bf16 (x) ----------------
__global__ void k_convert_x(const float* __restrict__ src,
                            unsigned short* __restrict__ dst, int n) {
    int i = (blockIdx.x * blockDim.x + threadIdx.x) * 4;
    if (i + 3 < n) {
        float4 v = *(const float4*)(src + i);
        ushort4 o;
        o.x = f2bf(v.x); o.y = f2bf(v.y); o.z = f2bf(v.z); o.w = f2bf(v.w);
        *(ushort4*)(dst + i) = o;
    }
}

// ------- 2) tiled transpose + convert: src [z][R][C] f32 -> dst [z][C][R] bf16 -------
__global__ void k_transpose_convert(const float* __restrict__ src,
                                    unsigned short* __restrict__ dst,
                                    int R, int C) {
    __shared__ unsigned short tile[32][33];
    const int z = blockIdx.z;
    const float* s = src + (size_t)z * R * C;
    unsigned short* d = dst + (size_t)z * R * C;
    const int c0 = blockIdx.x * 32, r0 = blockIdx.y * 32;
    const int tx = threadIdx.x, ty = threadIdx.y;            // 32 x 8
#pragma unroll
    for (int i = 0; i < 32; i += 8)
        tile[ty + i][tx] = f2bf(s[(size_t)(r0 + ty + i) * C + (c0 + tx)]);
    __syncthreads();
#pragma unroll
    for (int i = 0; i < 32; i += 8)
        d[(size_t)(c0 + ty + i) * R + (r0 + tx)] = tile[tx][ty + i];
}

// ---------------- 3a) gate logits: [NTOK][8] = x @ gate_w + gate_b ----------------
__global__ void k_gate(const float* __restrict__ x,
                       const float* __restrict__ gw,
                       const float* __restrict__ gb,
                       float* __restrict__ logits) {
    int id = blockIdx.x * blockDim.x + threadIdx.x;          // NTOK*8 threads
    if (id >= NTOK * N_EXP) return;
    int t = id >> 3, e = id & 7;
    const float* xr = x + (size_t)t * D_MODEL;
    float acc = gb[e];
    for (int d = 0; d < D_MODEL; ++d) acc = fmaf(xr[d], gw[d * N_EXP + e], acc);
    logits[id] = acc;
}

// ---- 3b) per-(batch,expert) column stats of softmax(axis=seq): m + log(sum exp) ----
__global__ void k_colstats(const float* __restrict__ logits, float* __restrict__ cst) {
    __shared__ float red[256];
    const int col = blockIdx.x;                  // 0..31  (b*8+e)
    const int b = col >> 3, e = col & 7;
    const float* p = logits + (size_t)b * SEQ * N_EXP + e;
    float m = -3.4e38f;
    for (int l = threadIdx.x; l < SEQ; l += 256) m = fmaxf(m, p[(size_t)l * N_EXP]);
    red[threadIdx.x] = m; __syncthreads();
    for (int s = 128; s > 0; s >>= 1) {
        if (threadIdx.x < s) red[threadIdx.x] = fmaxf(red[threadIdx.x], red[threadIdx.x + s]);
        __syncthreads();
    }
    m = red[0]; __syncthreads();
    float acc = 0.f;
    for (int l = threadIdx.x; l < SEQ; l += 256) acc += expf(p[(size_t)l * N_EXP] - m);
    red[threadIdx.x] = acc; __syncthreads();
    for (int s = 128; s > 0; s >>= 1) {
        if (threadIdx.x < s) red[threadIdx.x] += red[threadIdx.x + s];
        __syncthreads();
    }
    if (threadIdx.x == 0) cst[col] = m + logf(red[0]);
}

__global__ void k_zero(int* p, int n) {
    int i = blockIdx.x * blockDim.x + threadIdx.x;
    if (i < n) p[i] = 0;
}

// ---- 3c) per-token argmax over experts (first max wins) + expert bucket build ----
__global__ void k_route(const float* __restrict__ logits,
                        const float* __restrict__ cst,
                        int* __restrict__ counts, int* __restrict__ perm) {
    int t = blockIdx.x * blockDim.x + threadIdx.x;
    if (t >= NTOK) return;
    int b = t >> 11;                              // t / SEQ
    float best = -3.4e38f; int bi = 0;
#pragma unroll
    for (int e = 0; e < N_EXP; ++e) {
        float v = logits[(size_t)t * N_EXP + e] - cst[b * N_EXP + e];
        if (v > best) { best = v; bi = e; }       // strict > keeps lowest index on ties
    }
    int slot = atomicAdd(&counts[bi], 1);
    perm[bi * NTOK + slot] = t;
}

// ---------------- 4a) GEMM1: h = gelu(x_g @ W1[e] + b1[e]), bf16 out ----------------
// one wave -> 32x32 output tile via 2x2 register blocking (4 wmma / 8 loads per chunk)
// block = 4 waves -> 32 tokens x 128 ffn cols
__global__ void __launch_bounds__(128)
k_ffn1(const unsigned short* __restrict__ xb,     // [NTOK][1024] bf16
       const unsigned short* __restrict__ w1t,    // [8][4096][1024] bf16 (transposed)
       const float* __restrict__ bias1,           // [8][4096]
       const int* __restrict__ counts, const int* __restrict__ perm,
       unsigned short* __restrict__ h) {          // [NTOK][4096] bf16
    const int e   = blockIdx.z;
    const int cnt = counts[e];
    const int m0  = blockIdx.y * 32;
    if (m0 >= cnt) return;                        // uniform: EXEC stays all-ones
    const int wave  = threadIdx.x >> 5;
    const int lane  = threadIdx.x & 31;
    const int col16 = lane & 15;
    const int kgrp  = lane >> 4;
    const int n0    = blockIdx.x * 128 + wave * 32;

    // clamped gathers for the two M sub-tiles (tail tiles compute garbage, store masked)
    int r0 = m0 + col16;      if (r0 >= cnt) r0 = cnt - 1;
    int r1 = m0 + 16 + col16; if (r1 >= cnt) r1 = cnt - 1;
    const unsigned short* a0 = xb + (size_t)perm[e * NTOK + r0] * D_MODEL;
    const unsigned short* a1 = xb + (size_t)perm[e * NTOK + r1] * D_MODEL;
    const unsigned short* bp0 = w1t + ((size_t)e * FFN + (n0 + col16)) * D_MODEL;
    const unsigned short* bp1 = w1t + ((size_t)e * FFN + (n0 + 16 + col16)) * D_MODEL;

    v8f acc00 = {}, acc01 = {}, acc10 = {}, acc11 = {};
    for (int k0 = 0; k0 < D_MODEL; k0 += 32) {
        FragU A0, A1, B0, B1;
        // A 16x32 bf16: lane half g holds K {k0+8g..+7} and {k0+16+8g..+7}
        A0.u[0] = *(const uint4*)(a0 + k0 + kgrp * 8);
        A0.u[1] = *(const uint4*)(a0 + k0 + 16 + kgrp * 8);
        A1.u[0] = *(const uint4*)(a1 + k0 + kgrp * 8);
        A1.u[1] = *(const uint4*)(a1 + k0 + 16 + kgrp * 8);
        // B 32x16 bf16: lane half g holds K {k0+8g..+7} and {k0+16+8g..+7}
        B0.u[0] = *(const uint4*)(bp0 + k0 + kgrp * 8);
        B0.u[1] = *(const uint4*)(bp0 + k0 + 16 + kgrp * 8);
        B1.u[0] = *(const uint4*)(bp1 + k0 + kgrp * 8);
        B1.u[1] = *(const uint4*)(bp1 + k0 + 16 + kgrp * 8);
        acc00 = wmma_bf16(A0.v, B0.v, acc00);
        acc01 = wmma_bf16(A0.v, B1.v, acc01);
        acc10 = wmma_bf16(A1.v, B0.v, acc10);
        acc11 = wmma_bf16(A1.v, B1.v, acc11);
    }
    const float bs0 = bias1[(size_t)e * FFN + n0 + col16];
    const float bs1 = bias1[(size_t)e * FFN + n0 + 16 + col16];
#pragma unroll
    for (int j = 0; j < 8; ++j) {                 // D: VGPR j -> row j (+8 for hi half)
        int row = m0 + j + 8 * kgrp;
        if (row < cnt) {
            int tok = perm[e * NTOK + row];
            unsigned short* hr = h + (size_t)tok * FFN;
            hr[n0 + col16]      = f2bf(gelu_exact(acc00[j] + bs0));
            hr[n0 + 16 + col16] = f2bf(gelu_exact(acc01[j] + bs1));
        }
        int row2 = row + 16;
        if (row2 < cnt) {
            int tok = perm[e * NTOK + row2];
            unsigned short* hr = h + (size_t)tok * FFN;
            hr[n0 + col16]      = f2bf(gelu_exact(acc10[j] + bs0));
            hr[n0 + 16 + col16] = f2bf(gelu_exact(acc11[j] + bs1));
        }
    }
}

// ---------------- 4b) GEMM2: out = h_g @ W2[e] + b2[e], f32 scatter ----------------
__global__ void __launch_bounds__(128)
k_ffn2(const unsigned short* __restrict__ h,      // [NTOK][4096] bf16
       const unsigned short* __restrict__ w2t,    // [8][1024][4096] bf16 (transposed)
       const float* __restrict__ bias2,           // [8][1024]
       const int* __restrict__ counts, const int* __restrict__ perm,
       float* __restrict__ out) {                 // [NTOK][1024] f32
    const int e   = blockIdx.z;
    const int cnt = counts[e];
    const int m0  = blockIdx.y * 32;
    if (m0 >= cnt) return;
    const int wave  = threadIdx.x >> 5;
    const int lane  = threadIdx.x & 31;
    const int col16 = lane & 15;
    const int kgrp  = lane >> 4;
    const int n0    = blockIdx.x * 128 + wave * 32;

    int r0 = m0 + col16;      if (r0 >= cnt) r0 = cnt - 1;
    int r1 = m0 + 16 + col16; if (r1 >= cnt) r1 = cnt - 1;
    const unsigned short* a0 = h + (size_t)perm[e * NTOK + r0] * FFN;
    const unsigned short* a1 = h + (size_t)perm[e * NTOK + r1] * FFN;
    const unsigned short* bp0 = w2t + ((size_t)e * D_MODEL + (n0 + col16)) * FFN;
    const unsigned short* bp1 = w2t + ((size_t)e * D_MODEL + (n0 + 16 + col16)) * FFN;

    v8f acc00 = {}, acc01 = {}, acc10 = {}, acc11 = {};
    for (int k0 = 0; k0 < FFN; k0 += 32) {
        FragU A0, A1, B0, B1;
        A0.u[0] = *(const uint4*)(a0 + k0 + kgrp * 8);
        A0.u[1] = *(const uint4*)(a0 + k0 + 16 + kgrp * 8);
        A1.u[0] = *(const uint4*)(a1 + k0 + kgrp * 8);
        A1.u[1] = *(const uint4*)(a1 + k0 + 16 + kgrp * 8);
        B0.u[0] = *(const uint4*)(bp0 + k0 + kgrp * 8);
        B0.u[1] = *(const uint4*)(bp0 + k0 + 16 + kgrp * 8);
        B1.u[0] = *(const uint4*)(bp1 + k0 + kgrp * 8);
        B1.u[1] = *(const uint4*)(bp1 + k0 + 16 + kgrp * 8);
        acc00 = wmma_bf16(A0.v, B0.v, acc00);
        acc01 = wmma_bf16(A0.v, B1.v, acc01);
        acc10 = wmma_bf16(A1.v, B0.v, acc10);
        acc11 = wmma_bf16(A1.v, B1.v, acc11);
    }
    const float bs0 = bias2[(size_t)e * D_MODEL + n0 + col16];
    const float bs1 = bias2[(size_t)e * D_MODEL + n0 + 16 + col16];
#pragma unroll
    for (int j = 0; j < 8; ++j) {
        int row = m0 + j + 8 * kgrp;
        if (row < cnt) {
            int tok = perm[e * NTOK + row];
            float* orow = out + (size_t)tok * D_MODEL;
            orow[n0 + col16]      = acc00[j] + bs0;
            orow[n0 + 16 + col16] = acc01[j] + bs1;
        }
        int row2 = row + 16;
        if (row2 < cnt) {
            int tok = perm[e * NTOK + row2];
            float* orow = out + (size_t)tok * D_MODEL;
            orow[n0 + col16]      = acc10[j] + bs0;
            orow[n0 + 16 + col16] = acc11[j] + bs1;
        }
    }
}

// ---------------- launcher ----------------
extern "C" void kernel_launch(void* const* d_in, const int* in_sizes, int n_in,
                              void* d_out, int out_size, void* d_ws, size_t ws_size,
                              hipStream_t stream) {
    const float* x      = (const float*)d_in[0];
    const float* gate_w = (const float*)d_in[1];
    const float* gate_b = (const float*)d_in[2];
    const float* W1     = (const float*)d_in[3];
    const float* b1     = (const float*)d_in[4];
    const float* W2     = (const float*)d_in[5];
    const float* b2     = (const float*)d_in[6];
    float* out = (float*)d_out;

    // workspace carve-up (~209 MB)
    unsigned short* xb  = (unsigned short*)d_ws;                 // 16 MB
    unsigned short* w1t = xb  + (size_t)NTOK * D_MODEL;          // 64 MB
    unsigned short* w2t = w1t + (size_t)N_EXP * D_MODEL * FFN;   // 64 MB
    unsigned short* h   = w2t + (size_t)N_EXP * FFN * D_MODEL;   // 64 MB
    float* logits = (float*)(h + (size_t)NTOK * FFN);            // 256 KB
    float* cst    = logits + (size_t)NTOK * N_EXP;               // 128 B
    int*   counts = (int*)(cst + BS * N_EXP);
    int*   perm   = counts + 16;                                 // 256 KB

    // precision conversion + weight transposes (B operand wants contiguous K)
    k_convert_x<<<(NTOK * D_MODEL) / (256 * 4), 256, 0, stream>>>(x, xb, NTOK * D_MODEL);
    k_transpose_convert<<<dim3(FFN / 32, D_MODEL / 32, N_EXP), dim3(32, 8), 0, stream>>>(
        W1, w1t, D_MODEL, FFN);
    k_transpose_convert<<<dim3(D_MODEL / 32, FFN / 32, N_EXP), dim3(32, 8), 0, stream>>>(
        W2, w2t, FFN, D_MODEL);

    // gating: logits -> seq-axis softmax stats -> per-token argmax + buckets
    k_gate<<<(NTOK * N_EXP) / 256, 256, 0, stream>>>(x, gate_w, gate_b, logits);
    k_colstats<<<BS * N_EXP, 256, 0, stream>>>(logits, cst);
    k_zero<<<1, 32, 0, stream>>>(counts, N_EXP);
    k_route<<<NTOK / 256, 256, 0, stream>>>(logits, cst, counts, perm);

    // expert FFN as dense per-expert WMMA GEMMs over gathered tokens
    // wave = 32x32 tile (2x2 wmma register blocking), block = 32 tokens x 128 cols
    k_ffn1<<<dim3(FFN / 128, NTOK / 32, N_EXP), 128, 0, stream>>>(
        xb, w1t, b1, counts, perm, h);
    k_ffn2<<<dim3(D_MODEL / 128, NTOK / 32, N_EXP), 128, 0, stream>>>(
        h, w2t, b2, counts, perm, out);
}
